// MultiHeadAttention_36009005810143
// MI455X (gfx1250) — matrix operand order repaired and checked by reading, not compiled, not run on hardware
//
#include <hip/hip_runtime.h>
#include <hip/hip_bf16.h>

// ---------------------------------------------------------------------------
// MHA forward for MI455X (gfx1250, wave32, WMMA 16x16x32 f16, f32 accum)
// B=8, T=1024, D=U=640, H=10, d=64
// ---------------------------------------------------------------------------

typedef _Float16 v8h  __attribute__((ext_vector_type(8)));
typedef _Float16 v16h __attribute__((ext_vector_type(16)));
typedef float    v8f  __attribute__((ext_vector_type(8)));

#define WMMA_F16(a, b, c) \
  __builtin_amdgcn_wmma_f32_16x16x32_f16(false, (a), false, (b), (short)0, (c), false, false)

__device__ __forceinline__ v16h cat8(v8h lo, v8h hi) {
  v16h r;
#pragma unroll
  for (int i = 0; i < 8; ++i) { r[i] = lo[i]; r[8 + i] = hi[i]; }
  return r;
}

// ---------------------------------------------------------------------------
// f32 -> f16 elementwise convert
// ---------------------------------------------------------------------------
__global__ void k_f32_to_f16(const float* __restrict__ in, _Float16* __restrict__ out, int n) {
  int i = blockIdx.x * blockDim.x + threadIdx.x;
  if (i < n) out[i] = (_Float16)in[i];
}

// ---------------------------------------------------------------------------
// Projection GEMM: [8192 x 640] (xh, f16) @ [640 x 640] (wh, f16) -> f16
// Output layouts: transposed==0 -> [h][b][t][dv]   (Q, V)
//                 transposed==1 -> [h][b][dv][t]   (K^T)
// Block = 128 threads (4 waves); each wave computes a 16x64 strip.
// ---------------------------------------------------------------------------
__global__ __launch_bounds__(128)
void k_proj(const _Float16* __restrict__ xh, const _Float16* __restrict__ wh,
            _Float16* __restrict__ out, int transposed) {
  const int lane = threadIdx.x & 31;
  const int wave = threadIdx.x >> 5;
  const int hi   = lane >> 4;        // which 16-lane half
  const int ln   = lane & 15;
  const int kc   = hi * 8;           // A-layout K-chunk offset for this lane
  const int m0   = blockIdx.x * 64 + wave * 16;  // token rows
  const int n0   = blockIdx.y * 64;              // feature columns

  v8f acc[4] = {};

  // A operand: lane ln<16 holds row m0+ln, K chunks {kc..kc+7, kc+16..kc+23}
  const _Float16* aprow = xh + (size_t)(m0 + ln) * 640 + kc;

  for (int k0 = 0; k0 < 640; k0 += 32) {
    v16h a = cat8(*(const v8h*)(aprow + k0), *(const v8h*)(aprow + k0 + 16));
    // B operand: lane holds W column n0+ln, K in table order {kc..kc+7, kc+16..kc+23}
    const _Float16* bprow = wh + (size_t)k0 * 640 + n0 + ln;
#pragma unroll
    for (int nt = 0; nt < 4; ++nt) {
      v16h bm;
      for (int i = 0; i < 16; ++i) bm[i] = bprow[(size_t)(i < 8 ? kc + i : 8 + kc + i) * 640 + nt * 16];
      acc[nt] = WMMA_F16(a, bm, acc[nt]);
    }
  }

  // D layout: VGPR r, lane: (M = r + 8*hi, N = ln)
#pragma unroll
  for (int nt = 0; nt < 4; ++nt) {
#pragma unroll
    for (int r = 0; r < 8; ++r) {
      const int tok  = m0 + r + kc;          // kc == 8*hi
      const int ncol = n0 + nt * 16 + ln;
      const int h  = ncol >> 6, dv = ncol & 63;
      const int bb = tok >> 10, t  = tok & 1023;
      const int hb = h * 8 + bb;
      const _Float16 val = (_Float16)acc[nt][r];
      if (transposed) out[(size_t)(hb * 64 + dv) * 1024 + t] = val;
      else            out[(size_t)hb * 65536 + (size_t)t * 64 + dv] = val;
    }
  }
}

// ---------------------------------------------------------------------------
// Per-(h,b,t) query/key padding masks: sign(sum |row|)
// ---------------------------------------------------------------------------
__global__ void k_masks(const _Float16* __restrict__ Qh, const _Float16* __restrict__ Kth,
                        float* __restrict__ qm, float* __restrict__ km) {
  int i = blockIdx.x * blockDim.x + threadIdx.x;
  if (i >= 10 * 8 * 1024) return;
  const int hb = i >> 10, t = i & 1023;
  float sq = 0.f, sk = 0.f;
#pragma unroll 8
  for (int d = 0; d < 64; ++d) {
    sq += fabsf((float)Qh[(size_t)i * 64 + d]);
    sk += fabsf((float)Kth[(size_t)(hb * 64 + d) * 1024 + t]);
  }
  qm[i] = sq > 0.f ? 1.f : 0.f;
  km[i] = sk > 0.f ? 1.f : 0.f;
}

// ---------------------------------------------------------------------------
// Fused flash attention: one block = 64 query rows of one (h,b);
// each wave owns 16 rows; online softmax over key tiles of 32.
// ---------------------------------------------------------------------------
__global__ __launch_bounds__(128)
void k_attn(const _Float16* __restrict__ Qh, const _Float16* __restrict__ Kth,
            const _Float16* __restrict__ Vh, const float* __restrict__ km,
            const float* __restrict__ qm, const int* __restrict__ maskI,
            float* __restrict__ out) {
  __shared__ _Float16 plds[4][16][32];   // per-wave 1KB region for P transpose

  const int lane = threadIdx.x & 31;
  const int wave = threadIdx.x >> 5;
  const int hi   = lane >> 4;
  const int ln   = lane & 15;
  const int kc   = hi * 8;
  const int hb   = blockIdx.y;            // h*8 + b
  const int h    = hb >> 3, bb = hb & 7;
  const int qbase = blockIdx.x * 64 + wave * 16;

  const _Float16* Qb = Qh  + (size_t)hb * 65536;   // [t][64]
  const _Float16* Kb = Kth + (size_t)hb * 65536;   // [64][t]
  const _Float16* Vb = Vh  + (size_t)hb * 65536;   // [t][64]

  // Q A-operand tiles for dk 0..31 and 32..63 (loaded once)
  const _Float16* qp = Qb + (size_t)(qbase + ln) * 64 + kc;
  const v16h qa0 = cat8(*(const v8h*)(qp),      *(const v8h*)(qp + 16));
  const v16h qa1 = cat8(*(const v8h*)(qp + 32), *(const v8h*)(qp + 48));

  v8f acc[4] = {};
  float rm[8], rl[8];
#pragma unroll
  for (int r = 0; r < 8; ++r) { rm[r] = -3.0e38f; rl[r] = 0.f; }

  for (int kt = 0; kt < 1024; kt += 32) {
    // ---- scores tile S: 16 q-rows x 32 keys ----
    v8f s[2];
#pragma unroll
    for (int nt = 0; nt < 2; ++nt) {
      // B = K^T: lane holds key column kt+nt*16+ln, dk in table order (resp. +32)
      v16h b0, b1;
      for (int i = 0; i < 16; ++i) {
        const int dk = i < 8 ? kc + i : 8 + kc + i;
        b0[i] = Kb[(size_t)dk * 1024 + kt + nt * 16 + ln];
        b1[i] = Kb[(size_t)(32 + dk) * 1024 + kt + nt * 16 + ln];
      }
      v8f z = {};
      z = WMMA_F16(qa0, b0, z);
      z = WMMA_F16(qa1, b1, z);
      s[nt] = z;
    }

    // ---- scale + key padding mask + strict causal mask ----
    float nm[8];
#pragma unroll
    for (int r = 0; r < 8; ++r) nm[r] = -3.0e38f;
#pragma unroll
    for (int nt = 0; nt < 2; ++nt) {
      const int col = kt + nt * 16 + ln;            // per-lane constant column
      const float kmv = km[hb * 1024 + col];
#pragma unroll
      for (int r = 0; r < 8; ++r) {
        float v = s[nt][r] * 0.125f;                // 1/sqrt(64)
        if (kmv == 0.f) v = -4294967295.0f;         // NEG_BIG
        const int rowg = qbase + r + kc;
        if (rowg <= col) v += -10000.0f;            // allowed only row > col
        s[nt][r] = v;
        nm[r] = fmaxf(nm[r], v);
      }
    }

    // ---- row max across the 16-lane group (xor <16 stays in group) ----
#pragma unroll
    for (int off = 8; off >= 1; off >>= 1)
#pragma unroll
      for (int r = 0; r < 8; ++r)
        nm[r] = fmaxf(nm[r], __shfl_xor(nm[r], off, 32));

    // ---- online softmax update ----
    float sc[8], rs[8];
#pragma unroll
    for (int r = 0; r < 8; ++r) {
      const float mn = fmaxf(rm[r], nm[r]);
      sc[r] = __expf(rm[r] - mn);
      rm[r] = mn;
      rs[r] = 0.f;
    }
#pragma unroll
    for (int nt = 0; nt < 2; ++nt)
#pragma unroll
      for (int r = 0; r < 8; ++r) {
        const float p = __expf(s[nt][r] - rm[r]);
        s[nt][r] = p;
        rs[r] += p;
      }
#pragma unroll
    for (int off = 8; off >= 1; off >>= 1)
#pragma unroll
      for (int r = 0; r < 8; ++r)
        rs[r] += __shfl_xor(rs[r], off, 32);
#pragma unroll
    for (int r = 0; r < 8; ++r) rl[r] = rl[r] * sc[r] + rs[r];
#pragma unroll
    for (int dv = 0; dv < 4; ++dv)
#pragma unroll
      for (int r = 0; r < 8; ++r)
        acc[dv][r] *= sc[r];

    // ---- P: D-layout -> LDS -> A-layout (per-wave region, same-wave order) ----
#pragma unroll
    for (int nt = 0; nt < 2; ++nt)
#pragma unroll
      for (int r = 0; r < 8; ++r)
        plds[wave][r + kc][nt * 16 + ln] = (_Float16)s[nt][r];
    asm volatile("s_wait_dscnt 0" ::: "memory");
    const v16h pa = cat8(*(const v8h*)(&plds[wave][ln][kc]),
                         *(const v8h*)(&plds[wave][ln][kc + 16]));

    // ---- O += P @ V : B = V, lane holds column dv*16+ln, keys in table order ----
#pragma unroll
    for (int dv = 0; dv < 4; ++dv) {
      v16h vb;
      for (int i = 0; i < 16; ++i) vb[i] = Vb[(size_t)(kt + (i < 8 ? kc + i : 8 + kc + i)) * 64 + dv * 16 + ln];
      acc[dv] = WMMA_F16(pa, vb, acc[dv]);
    }
  }

  // ---- finalize: /l, query mask, input mask, store fp32 [B,T,U] ----
#pragma unroll
  for (int dv = 0; dv < 4; ++dv)
#pragma unroll
    for (int r = 0; r < 8; ++r) {
      const int rowg = qbase + r + kc;
      float o = acc[dv][r] / rl[r];
      o *= qm[hb * 1024 + rowg];
      o *= (float)maskI[bb * 1024 + rowg];
      out[(size_t)(bb * 1024 + rowg) * 640 + h * 64 + dv * 16 + ln] = o;
    }
}

// ---------------------------------------------------------------------------
// Launch
// ---------------------------------------------------------------------------
extern "C" void kernel_launch(void* const* d_in, const int* in_sizes, int n_in,
                              void* d_out, int out_size, void* d_ws, size_t ws_size,
                              hipStream_t stream) {
  (void)in_sizes; (void)n_in; (void)out_size; (void)ws_size;
  const float* x     = (const float*)d_in[0];
  const int*   maskI = (const int*)d_in[1];
  const float* Wq    = (const float*)d_in[2];
  const float* Wk    = (const float*)d_in[3];
  const float* Wv    = (const float*)d_in[4];
  float* out = (float*)d_out;

  char* ws = (char*)d_ws;
  size_t off = 0;
  auto take = [&](size_t bytes) -> char* {
    char* p = ws + off;
    off = (off + bytes + 255) & ~(size_t)255;
    return p;
  };
  _Float16* xh  = (_Float16*)take((size_t)8192 * 640 * 2);
  _Float16* whq = (_Float16*)take((size_t)640 * 640 * 2);
  _Float16* whk = (_Float16*)take((size_t)640 * 640 * 2);
  _Float16* whv = (_Float16*)take((size_t)640 * 640 * 2);
  _Float16* Qh  = (_Float16*)take((size_t)10 * 8 * 1024 * 64 * 2);  // [h,b,t,d]
  _Float16* Kth = (_Float16*)take((size_t)10 * 8 * 1024 * 64 * 2);  // [h,b,d,t]
  _Float16* Vh  = (_Float16*)take((size_t)10 * 8 * 1024 * 64 * 2);  // [h,b,t,d]
  float* qmv = (float*)take((size_t)10 * 8 * 1024 * 4);
  float* kmv = (float*)take((size_t)10 * 8 * 1024 * 4);

  // 1) f16 conversions
  k_f32_to_f16<<<(8192 * 640 + 255) / 256, 256, 0, stream>>>(x, xh, 8192 * 640);
  k_f32_to_f16<<<(640 * 640 + 255) / 256, 256, 0, stream>>>(Wq, whq, 640 * 640);
  k_f32_to_f16<<<(640 * 640 + 255) / 256, 256, 0, stream>>>(Wk, whk, 640 * 640);
  k_f32_to_f16<<<(640 * 640 + 255) / 256, 256, 0, stream>>>(Wv, whv, 640 * 640);

  // 2) projections (WMMA GEMMs)
  dim3 pg(128, 10);
  k_proj<<<pg, 128, 0, stream>>>(xh, whq, Qh, 0);
  k_proj<<<pg, 128, 0, stream>>>(xh, whk, Kth, 1);
  k_proj<<<pg, 128, 0, stream>>>(xh, whv, Vh, 0);

  // 3) padding masks
  k_masks<<<(10 * 8 * 1024 + 255) / 256, 256, 0, stream>>>(Qh, Kth, qmv, kmv);

  // 4) fused flash attention
  dim3 ag(16, 80);  // (T/64 q-blocks, H*B)
  k_attn<<<ag, 128, 0, stream>>>(Qh, Kth, Vh, kmv, qmv, maskI, out);
}
